// WhisperDecoderStep_87497073754620
// MI455X (gfx1250) — matrix operand order repaired and checked by reading, not compiled, not run on hardware
//
#include <hip/hip_runtime.h>
#include <math.h>

// ---------------------------------------------------------------------------
// Whisper decoder forward on gfx1250 (CDNA5).
// All GEMMs via v_wmma_f32_16x16x32_bf16 (bf16 operands, f32 accumulate).
// Staging uses 8-byte vector global loads + ds_store_b64 on interior tiles.
// ---------------------------------------------------------------------------

typedef __attribute__((ext_vector_type(16))) __bf16 v16bf;
typedef __attribute__((ext_vector_type(4)))  __bf16 v4bf;
typedef __attribute__((ext_vector_type(8)))  float  v8f;

#define TILE_M 64
#define TILE_N 64
#define TILE_K 64
#define LDS_K  (TILE_K + 8)   // A tile row stride (elems)
#define LDS_N  (TILE_N + 8)   // B tile row stride (elems)

// flags
#define GF_BT     1   // B given as [N,K] row-major (use B^T)
#define GF_GELU   2   // exact gelu epilogue
#define GF_CAUSAL 4   // scores: col > row -> -1e30

__device__ __forceinline__ __bf16 f2bf(float f) { return (__bf16)f; }

// A fragment: 16x32 bf16 from LDS tile [M][K] (row stride LDS_K), K offset ko.
// ISA 7.12.2: lanes 0-15 & 16-31 both cover M=0..15; lane group selects K
// ranges {0..7,16..23} vs {8..15,24..31}; VGPR v holds a K pair.
__device__ __forceinline__ v16bf load_a_frag(const __bf16* lds, int rb, int ko, int lane) {
  v16bf f;
  const __bf16* row = lds + (rb + (lane & 15)) * LDS_K + ko;
  int g = (lane >> 4) * 8;
#pragma unroll
  for (int v = 0; v < 8; ++v) {
    int kb = ((v < 4) ? 0 : 16) + g + (v & 3) * 2;
    f[2 * v]     = row[kb];
    f[2 * v + 1] = row[kb + 1];
  }
  return f;
}

// B fragment: 32x16 bf16 from LDS tile [K][N] (row stride LDS_N), K offset ko.
// Lane l holds column n=cb+(l&15); K order matches A: {0..7,16..23} / {8..15,24..31}.
__device__ __forceinline__ v16bf load_b_frag(const __bf16* lds, int cb, int ko, int lane) {
  v16bf f;
  const __bf16* col = lds + ko * LDS_N + cb + (lane & 15);
  int g = (lane >> 4) * 8;
#pragma unroll
  for (int i = 0; i < 16; ++i) f[i] = col[(((i < 8) ? 0 : 16) + g + (i & 7)) * LDS_N];
  return f;
}

__global__ __launch_bounds__(256) void wmma_gemm_kernel(
    const __bf16* __restrict__ A, int lda, long strideA,
    const __bf16* __restrict__ B, int ldb, long strideB,
    const float* __restrict__ bias,
    const float* __restrict__ residual,
    float* __restrict__ outF, __bf16* __restrict__ outB, int ldc, long strideC,
    int M, int N, int K, float scale, int flags)
{
  __shared__ __bf16 As[TILE_M * LDS_K];
  __shared__ __bf16 Bs[TILE_K * LDS_N];

  const int bz = blockIdx.z;
  A += (long)bz * strideA;
  B += (long)bz * strideB;
  const long coff = (long)bz * strideC;

  const int n0 = blockIdx.x * TILE_N;
  const int m0 = blockIdx.y * TILE_M;

  const int tid  = threadIdx.x;
  const int lane = tid & 31;
  const int wave = tid >> 5;
  const int rb   = (wave & 3) * 16;  // wave row within tile
  const int cb   = (wave >> 2) * 32; // wave col half within tile

  v8f acc0 = {0.f, 0.f, 0.f, 0.f, 0.f, 0.f, 0.f, 0.f};
  v8f acc1 = {0.f, 0.f, 0.f, 0.f, 0.f, 0.f, 0.f, 0.f};

  const bool bt     = (flags & GF_BT) != 0;
  const bool m_full = (m0 + TILE_M <= M);
  const bool n_full = (n0 + TILE_N <= N);

  for (int k0 = 0; k0 < K; k0 += TILE_K) {
    const bool k_full = (k0 + TILE_K <= K);

    // ---- stage A tile [64][64] ----
    if (m_full && k_full) {
      // 4 x 8-byte vector loads per thread (requires lda%4==0, 8B-aligned base)
#pragma unroll
      for (int i = 0; i < 4; ++i) {
        int idx = tid + i * 256;
        int r  = idx >> 4;
        int cg = (idx & 15) * 4;
        *(v4bf*)&As[r * LDS_K + cg] =
            *(const v4bf*)&A[(long)(m0 + r) * lda + (k0 + cg)];
      }
    } else {
#pragma unroll
      for (int i = 0; i < 16; ++i) {
        int idx = tid + i * 256;
        int r = idx >> 6, c = idx & 63;
        int gm = m0 + r, gk = k0 + c;
        int gmc = gm < M ? gm : M - 1;
        int gkc = gk < K ? gk : K - 1;
        __bf16 v = A[(long)gmc * lda + gkc];
        As[r * LDS_K + c] = (gm < M && gk < K) ? v : f2bf(0.f);
      }
    }

    // ---- stage B tile as [K=64][N=64] ----
    if (!bt) {
      if (k_full && n_full) {
#pragma unroll
        for (int i = 0; i < 4; ++i) {
          int idx = tid + i * 256;
          int r  = idx >> 4;
          int cg = (idx & 15) * 4;
          *(v4bf*)&Bs[r * LDS_N + cg] =
              *(const v4bf*)&B[(long)(k0 + r) * ldb + (n0 + cg)];
        }
      } else {
#pragma unroll
        for (int i = 0; i < 16; ++i) {
          int idx = tid + i * 256;
          int r = idx >> 6, c = idx & 63;
          int gk = k0 + r, gn = n0 + c;
          int gkc = gk < K ? gk : K - 1;
          int gnc = gn < N ? gn : N - 1;
          __bf16 v = B[(long)gkc * ldb + gnc];
          Bs[r * LDS_N + c] = (gk < K && gn < N) ? v : f2bf(0.f);
        }
      }
    } else {
      // B is [N,K]: transpose into [K][N] while staging
      if (k_full && n_full) {
#pragma unroll
        for (int i = 0; i < 4; ++i) {
          int idx = tid + i * 256;
          int n  = idx >> 4;
          int kg = (idx & 15) * 4;
          v4bf t = *(const v4bf*)&B[(long)(n0 + n) * ldb + (k0 + kg)];
#pragma unroll
          for (int j = 0; j < 4; ++j) Bs[(kg + j) * LDS_N + n] = t[j];
        }
      } else {
#pragma unroll
        for (int i = 0; i < 16; ++i) {
          int idx = tid + i * 256;
          int n = idx >> 6, kk = idx & 63;
          int gn = n0 + n, gk = k0 + kk;
          int gnc = gn < N ? gn : N - 1;
          int gkc = gk < K ? gk : K - 1;
          __bf16 v = B[(long)gnc * ldb + gkc];
          Bs[kk * LDS_N + n] = (gn < N && gk < K) ? v : f2bf(0.f);
        }
      }
    }
    __syncthreads();

    v16bf a0 = load_a_frag(As, rb, 0,  lane);
    v16bf a1 = load_a_frag(As, rb, 32, lane);

    v16bf b00 = load_b_frag(Bs, cb, 0,  lane);
    v16bf b01 = load_b_frag(Bs, cb, 32, lane);
    acc0 = __builtin_amdgcn_wmma_f32_16x16x32_bf16(false, a0, false, b00, (short)0, acc0, false, false);
    acc0 = __builtin_amdgcn_wmma_f32_16x16x32_bf16(false, a1, false, b01, (short)0, acc0, false, false);

    v16bf b10 = load_b_frag(Bs, cb + 16, 0,  lane);
    v16bf b11 = load_b_frag(Bs, cb + 16, 32, lane);
    acc1 = __builtin_amdgcn_wmma_f32_16x16x32_bf16(false, a0, false, b10, (short)0, acc1, false, false);
    acc1 = __builtin_amdgcn_wmma_f32_16x16x32_bf16(false, a1, false, b11, (short)0, acc1, false, false);

    __syncthreads();
  }

  // epilogue: C layout — VGPR r: lanes 0-15 -> M=r, lanes 16-31 -> M=8+r; N=lane&15
  const int mbase = m0 + rb + ((lane >> 4) << 3);
  const int nb0   = n0 + cb + (lane & 15);
#pragma unroll
  for (int r = 0; r < 8; ++r) {
    int gm = mbase + r;
    if (gm >= M) continue;
#pragma unroll
    for (int half = 0; half < 2; ++half) {
      int gn = nb0 + half * 16;
      if (gn >= N) continue;
      float v = (half ? acc1[r] : acc0[r]) * scale;
      if (bias) v += bias[gn];
      if (flags & GF_GELU) v = 0.5f * v * (1.0f + erff(v * 0.70710678118654752f));
      if ((flags & GF_CAUSAL) && gn > gm) v = -1e30f;
      long o = coff + (long)gm * ldc + gn;
      if (residual) v += residual[o];
      if (outF) outF[o] = v;
      if (outB) outB[o] = f2bf(v);
    }
  }
}

// ---------------------------------------------------------------------------
__global__ __launch_bounds__(256) void cvt_f32_bf16_kernel(
    const float* __restrict__ src, __bf16* __restrict__ dst, long n)
{
  long i = (long)blockIdx.x * 256 + threadIdx.x;
  long stride = (long)gridDim.x * 256;
  for (; i < n; i += stride) dst[i] = f2bf(src[i]);
}

__global__ __launch_bounds__(256) void embed_kernel(
    const int* __restrict__ tokens, const float* __restrict__ temb,
    const float* __restrict__ pemb, float* __restrict__ x, int S_, int D_)
{
  int row = blockIdx.x;      // b*S + s
  int s   = row % S_;
  int tok = tokens[row];
  const float* te = temb + (long)tok * D_;
  const float* pe = pemb + (long)s * D_;
  float* xr = x + (long)row * D_;
  for (int d = threadIdx.x; d < D_; d += 256) xr[d] = te[d] + pe[d];
}

__global__ __launch_bounds__(256) void layernorm_kernel(
    const float* __restrict__ x, const float* __restrict__ w,
    const float* __restrict__ b, __bf16* __restrict__ out, int D_)
{
  __shared__ float red[256];
  const int row = blockIdx.x;
  const int tid = threadIdx.x;
  const float* xr = x + (long)row * D_;

  float s = 0.f;
  for (int d = tid; d < D_; d += 256) s += xr[d];
  red[tid] = s; __syncthreads();
  for (int st = 128; st > 0; st >>= 1) { if (tid < st) red[tid] += red[tid + st]; __syncthreads(); }
  float mean = red[0] / (float)D_;
  __syncthreads();

  float vs = 0.f;
  for (int d = tid; d < D_; d += 256) { float t = xr[d] - mean; vs += t * t; }
  red[tid] = vs; __syncthreads();
  for (int st = 128; st > 0; st >>= 1) { if (tid < st) red[tid] += red[tid + st]; __syncthreads(); }
  float inv = rsqrtf(red[0] / (float)D_ + 1e-5f);
  __syncthreads();

  __bf16* orow = out + (long)row * D_;
  for (int d = tid; d < D_; d += 256)
    orow[d] = f2bf((xr[d] - mean) * inv * w[d] + b[d]);
}

__global__ __launch_bounds__(256) void softmax_kernel(
    const float* __restrict__ sc, __bf16* __restrict__ pr, int T)
{
  __shared__ float red[256];
  const int row = blockIdx.x;
  const int tid = threadIdx.x;
  const float* sr = sc + (long)row * T;

  float mx = -1e30f;
  for (int t = tid; t < T; t += 256) mx = fmaxf(mx, sr[t]);
  red[tid] = mx; __syncthreads();
  for (int st = 128; st > 0; st >>= 1) { if (tid < st) red[tid] = fmaxf(red[tid], red[tid + st]); __syncthreads(); }
  mx = red[0]; __syncthreads();

  float sum = 0.f;
  for (int t = tid; t < T; t += 256) sum += __expf(sr[t] - mx);
  red[tid] = sum; __syncthreads();
  for (int st = 128; st > 0; st >>= 1) { if (tid < st) red[tid] += red[tid + st]; __syncthreads(); }
  float inv = 1.0f / red[0];

  __bf16* prow = pr + (long)row * T;
  for (int t = tid; t < T; t += 256) prow[t] = f2bf(__expf(sr[t] - mx) * inv);
}

// ---------------------------------------------------------------------------
extern "C" void kernel_launch(void* const* d_in, const int* in_sizes, int n_in,
                              void* d_out, int out_size, void* d_ws, size_t ws_size,
                              hipStream_t stream)
{
  (void)in_sizes; (void)n_in; (void)out_size; (void)ws_size;
  constexpr int Lc = 6, Dc = 768, Hc = 12, FFc = 3072, Vc = 51865;
  constexpr int Ac = 1500, Bc = 32, Sc = 448, DHc = 64;
  constexpr int BS = Bc * Sc;

  const int*   tokens = (const int*)d_in[0];
  const float* audio  = (const float*)d_in[1];
  const float* temb   = (const float*)d_in[2];
  const float* pemb   = (const float*)d_in[3];
  const float* attn_ln_w = (const float*)d_in[4];
  const float* attn_ln_b = (const float*)d_in[5];
  const float* Wq = (const float*)d_in[6];  const float* bq = (const float*)d_in[7];
  const float* Wk = (const float*)d_in[8];
  const float* Wv = (const float*)d_in[9];  const float* bv = (const float*)d_in[10];
  const float* Wo = (const float*)d_in[11]; const float* bo = (const float*)d_in[12];
  const float* cross_ln_w = (const float*)d_in[13];
  const float* cross_ln_b = (const float*)d_in[14];
  const float* cWq = (const float*)d_in[15]; const float* cbq = (const float*)d_in[16];
  const float* cWk = (const float*)d_in[17];
  const float* cWv = (const float*)d_in[18]; const float* cbv = (const float*)d_in[19];
  const float* cWo = (const float*)d_in[20]; const float* cbo = (const float*)d_in[21];
  const float* mlp_ln_w = (const float*)d_in[22];
  const float* mlp_ln_b = (const float*)d_in[23];
  const float* W1 = (const float*)d_in[24]; const float* b1 = (const float*)d_in[25];
  const float* W2 = (const float*)d_in[26]; const float* b2 = (const float*)d_in[27];
  const float* ln_w = (const float*)d_in[28];
  const float* ln_b = (const float*)d_in[29];

  // ---- workspace carve-up ----
  size_t off = 0;
  auto take = [&](size_t bytes) -> char* {
    char* r = (char*)d_ws + off;
    off = (off + bytes + 255) & ~(size_t)255;
    return r;
  };
  const long szDD = (long)Lc * Dc * Dc;
  __bf16* bWq  = (__bf16*)take(szDD * 2);
  __bf16* bWk  = (__bf16*)take(szDD * 2);
  __bf16* bWv  = (__bf16*)take(szDD * 2);
  __bf16* bWo  = (__bf16*)take(szDD * 2);
  __bf16* bcWq = (__bf16*)take(szDD * 2);
  __bf16* bcWk = (__bf16*)take(szDD * 2);
  __bf16* bcWv = (__bf16*)take(szDD * 2);
  __bf16* bcWo = (__bf16*)take(szDD * 2);
  __bf16* bW1  = (__bf16*)take((long)Lc * Dc * FFc * 2);
  __bf16* bW2  = (__bf16*)take((long)Lc * FFc * Dc * 2);
  __bf16* bEmb = (__bf16*)take((long)Vc * Dc * 2);
  __bf16* bAf  = (__bf16*)take((long)Bc * Ac * Dc * 2);
  float*  xf   = (float*) take((long)BS * Dc * 4);
  __bf16* hb   = (__bf16*)take((long)BS * Dc * 2);
  __bf16* Qb   = (__bf16*)take((long)BS * Dc * 2);
  __bf16* Kb   = (__bf16*)take((long)BS * Dc * 2);
  __bf16* Vb   = (__bf16*)take((long)BS * Dc * 2);
  __bf16* att  = (__bf16*)take((long)BS * Dc * 2);
  __bf16* cKb  = (__bf16*)take((long)Bc * Ac * Dc * 2);
  __bf16* cVb  = (__bf16*)take((long)Bc * Ac * Dc * 2);
  __bf16* mlpt = (__bf16*)take((long)BS * FFc * 2);
  float*  scor = (float*) take((long)Hc * Sc * Ac * 4);
  __bf16* prob = (__bf16*)take((long)Hc * Sc * Ac * 2);

  auto cvt = [&](const float* s, __bf16* d, long n) {
    cvt_f32_bf16_kernel<<<2048, 256, 0, stream>>>(s, d, n);
  };
  auto gemm = [&](const __bf16* A, int lda, long sA,
                  const __bf16* B, int ldb, long sB,
                  const float* bias, const float* res,
                  float* oF, __bf16* oB, int ldc, long sC,
                  int M, int N, int K, float scale, int flags, int nz) {
    dim3 g((N + TILE_N - 1) / TILE_N, (M + TILE_M - 1) / TILE_M, nz);
    wmma_gemm_kernel<<<g, 256, 0, stream>>>(A, lda, sA, B, ldb, sB, bias, res,
                                            oF, oB, ldc, sC, M, N, K, scale, flags);
  };
  auto ln = [&](const float* x, const float* w, const float* b, __bf16* o, int rows) {
    layernorm_kernel<<<rows, 256, 0, stream>>>(x, w, b, o, Dc);
  };

  // ---- weight / input conversion to bf16 ----
  cvt(Wq, bWq, szDD);   cvt(Wk, bWk, szDD);   cvt(Wv, bWv, szDD);   cvt(Wo, bWo, szDD);
  cvt(cWq, bcWq, szDD); cvt(cWk, bcWk, szDD); cvt(cWv, bcWv, szDD); cvt(cWo, bcWo, szDD);
  cvt(W1, bW1, (long)Lc * Dc * FFc);
  cvt(W2, bW2, (long)Lc * FFc * Dc);
  cvt(temb, bEmb, (long)Vc * Dc);
  cvt(audio, bAf, (long)Bc * Ac * Dc);

  // ---- embedding ----
  embed_kernel<<<BS, 256, 0, stream>>>(tokens, temb, pemb, xf, Sc, Dc);

  const float sc_qk = 0.125f; // DH^-0.5, DH=64

  for (int l = 0; l < Lc; ++l) {
    const __bf16* wq = bWq + (long)l * Dc * Dc;
    const __bf16* wk = bWk + (long)l * Dc * Dc;
    const __bf16* wv = bWv + (long)l * Dc * Dc;
    const __bf16* wo = bWo + (long)l * Dc * Dc;
    const __bf16* cwq = bcWq + (long)l * Dc * Dc;
    const __bf16* cwk = bcWk + (long)l * Dc * Dc;
    const __bf16* cwv = bcWv + (long)l * Dc * Dc;
    const __bf16* cwo = bcWo + (long)l * Dc * Dc;
    const __bf16* w1 = bW1 + (long)l * Dc * FFc;
    const __bf16* w2 = bW2 + (long)l * FFc * Dc;

    // ===== self attention =====
    ln(xf, attn_ln_w + l * Dc, attn_ln_b + l * Dc, hb, BS);
    gemm(hb, Dc, 0, wq, Dc, 0, bq + l * Dc, nullptr, nullptr, Qb, Dc, 0, BS, Dc, Dc, 1.f, 0, 1);
    gemm(hb, Dc, 0, wk, Dc, 0, nullptr,     nullptr, nullptr, Kb, Dc, 0, BS, Dc, Dc, 1.f, 0, 1);
    gemm(hb, Dc, 0, wv, Dc, 0, bv + l * Dc, nullptr, nullptr, Vb, Dc, 0, BS, Dc, Dc, 1.f, 0, 1);

    for (int b = 0; b < Bc; ++b) {
      const __bf16* Qbb = Qb + (long)b * Sc * Dc;
      const __bf16* Kbb = Kb + (long)b * Sc * Dc;
      const __bf16* Vbb = Vb + (long)b * Sc * Dc;
      // scores[h,s,t] = scale * Q . K^T  (+ causal mask)
      gemm(Qbb, Dc, DHc, Kbb, Dc, DHc, nullptr, nullptr, scor, nullptr,
           Sc, (long)Sc * Sc, Sc, Sc, DHc, sc_qk, GF_BT | GF_CAUSAL, Hc);
      softmax_kernel<<<Hc * Sc, 256, 0, stream>>>(scor, prob, Sc);
      // out[h-slice] = P @ V
      gemm(prob, Sc, (long)Sc * Sc, Vbb, Dc, DHc, nullptr, nullptr, nullptr,
           att + (long)b * Sc * Dc, Dc, DHc, Sc, DHc, Sc, 1.f, 0, Hc);
    }
    // x += att @ Wo + bo
    gemm(att, Dc, 0, wo, Dc, 0, bo + l * Dc, xf, xf, nullptr, Dc, 0, BS, Dc, Dc, 1.f, 0, 1);

    // ===== cross attention =====
    ln(xf, cross_ln_w + l * Dc, cross_ln_b + l * Dc, hb, BS);
    gemm(hb, Dc, 0, cwq, Dc, 0, cbq + l * Dc, nullptr, nullptr, Qb, Dc, 0, BS, Dc, Dc, 1.f, 0, 1);
    gemm(bAf, Dc, 0, cwk, Dc, 0, nullptr,      nullptr, nullptr, cKb, Dc, 0, Bc * Ac, Dc, Dc, 1.f, 0, 1);
    gemm(bAf, Dc, 0, cwv, Dc, 0, cbv + l * Dc, nullptr, nullptr, cVb, Dc, 0, Bc * Ac, Dc, Dc, 1.f, 0, 1);

    for (int b = 0; b < Bc; ++b) {
      const __bf16* Qbb = Qb  + (long)b * Sc * Dc;
      const __bf16* Kbb = cKb + (long)b * Ac * Dc;
      const __bf16* Vbb = cVb + (long)b * Ac * Dc;
      gemm(Qbb, Dc, DHc, Kbb, Dc, DHc, nullptr, nullptr, scor, nullptr,
           Ac, (long)Sc * Ac, Sc, Ac, DHc, sc_qk, GF_BT, Hc);
      softmax_kernel<<<Hc * Sc, 256, 0, stream>>>(scor, prob, Ac);
      gemm(prob, Ac, (long)Sc * Ac, Vbb, Dc, DHc, nullptr, nullptr, nullptr,
           att + (long)b * Sc * Dc, Dc, DHc, Sc, DHc, Ac, 1.f, 0, Hc);
    }
    gemm(att, Dc, 0, cwo, Dc, 0, cbo + l * Dc, xf, xf, nullptr, Dc, 0, BS, Dc, Dc, 1.f, 0, 1);

    // ===== MLP =====
    ln(xf, mlp_ln_w + l * Dc, mlp_ln_b + l * Dc, hb, BS);
    gemm(hb, Dc, 0, w1, FFc, 0, b1 + l * FFc, nullptr, nullptr, mlpt, FFc, 0,
         BS, FFc, Dc, 1.f, GF_GELU, 1);
    gemm(mlpt, FFc, 0, w2, Dc, 0, b2 + l * Dc, xf, xf, nullptr, Dc, 0,
         BS, Dc, FFc, 1.f, 0, 1);
  }

  // ===== final LN + logits = h @ token_emb^T =====
  ln(xf, ln_w, ln_b, hb, BS);
  gemm(hb, Dc, 0, bEmb, Dc, 0, nullptr, nullptr, (float*)d_out, nullptr, Vc, 0,
       BS, Vc, Dc, 1.f, GF_BT, 1);
}
